// MetabolicPathwayLoss_14096082665810
// MI455X (gfx1250) — matrix operand order repaired and checked by reading, not compiled, not run on hardware
//
#include <hip/hip_runtime.h>
#include <hip/hip_bf16.h>

// MI455X / gfx1250, wave32. BF16 WMMA with fp32 accumulate.
typedef __attribute__((ext_vector_type(16))) __bf16 v16bf;
typedef __attribute__((ext_vector_type(8)))  float  v8f;

#define N_NODES 8192
#define P_DIM   128
#define D_DIM   256

#define WMMA_BF16(A, B, C) \
    __builtin_amdgcn_wmma_f32_16x16x32_bf16(false, (A), false, (B), (short)0, (C), false, false)

union FragB16 { v16bf v; uint4 q[2]; };

__device__ __forceinline__ float waveReduce(float v) {
#pragma unroll
    for (int off = 16; off > 0; off >>= 1) v += __shfl_xor(v, off, 32);
    return v;
}

__global__ void init_out(float* out) { out[0] = 0.0f; }

// --- Prep 1: row-normalize embeddings -> bf16.  One wave per row of D=256.
__global__ void normalize_embeddings(const float* __restrict__ E,
                                     __bf16* __restrict__ Xn) {
    const int wave = threadIdx.x >> 5;
    const int lane = threadIdx.x & 31;
    const int row  = blockIdx.x * (blockDim.x >> 5) + wave;
    if (row >= N_NODES) return;
    const float* e = E + (size_t)row * D_DIM;
    float x[8];
    float s = 0.0f;
#pragma unroll
    for (int j = 0; j < 8; ++j) { x[j] = e[lane + 32 * j]; s += x[j] * x[j]; }
    s = waveReduce(s);
    const float scale = 1.0f / fmaxf(sqrtf(s), 1e-8f);
    __bf16* o = Xn + (size_t)row * D_DIM;
#pragma unroll
    for (int j = 0; j < 8; ++j) o[lane + 32 * j] = (__bf16)(x[j] * scale);
}

// --- Prep 2: A -> bf16 (row-major + transposed), fused weight loss.
__global__ void convert_and_weight(const float* __restrict__ A,
                                   const float* __restrict__ W,
                                   __bf16* __restrict__ Abf,
                                   __bf16* __restrict__ AbfT,
                                   float* __restrict__ out) {
    const int tid    = blockIdx.x * blockDim.x + threadIdx.x;
    const int stride = gridDim.x * blockDim.x;
    float s = 0.0f;
    for (int i = tid; i < N_NODES * P_DIM; i += stride) {
        const float a = A[i];
        const int r = i >> 7;      // / P_DIM
        const int c = i & (P_DIM - 1);
        Abf[i] = (__bf16)a;
        AbfT[(size_t)c * N_NODES + r] = (__bf16)a;
        const float d = a - W[i];
        s += d * d;
    }
    s = waveReduce(s);
    if ((threadIdx.x & 31) == 0)
        atomicAdd(out, s * (1.0f / ((float)N_NODES * (float)P_DIM)));
}

// --- Coherence: fused  mean((A@A^T - Xn@Xn^T)^2)  over symmetric 32x32 blocks.
// One wave per 32x32 block (2x2 register-blocked 16x16 WMMA tiles): 4 fragment
// loads feed 4 WMMAs per K-step, halving L0 operand traffic vs 1x1 tiling.
// Only R <= C blocks computed; off-diagonal blocks weighted 2x.
__global__ void __launch_bounds__(128)
coherence_kernel(const __bf16* __restrict__ Abf,
                 const __bf16* __restrict__ Xn,
                 float* __restrict__ out) {
    const int lane = threadIdx.x & 31;
    const int wave = threadIdx.x >> 5;
    const int C = blockIdx.x * 4 + wave;    // 32-col block 0..255
    const int R = blockIdx.y;               // 32-row block 0..255
    if (R > C) return;

    const int m  = lane & 15;
    const int kA = (lane < 16) ? 0 : 8;     // A-operand K-chunk base
    const int kB = (lane < 16) ? 0 : 8;     // B-operand K-chunk base
    const int r0 = R * 32 + m, r1 = r0 + 16;
    const int c0 = C * 32 + m, c1 = c0 + 16;

    // S1 = A @ A^T 32x32 block, K = 128
    v8f p00 = {}, p01 = {}, p10 = {}, p11 = {};
    {
        const __bf16* a0 = Abf + (size_t)r0 * P_DIM;
        const __bf16* a1 = Abf + (size_t)r1 * P_DIM;
        const __bf16* b0 = Abf + (size_t)c0 * P_DIM;
        const __bf16* b1 = Abf + (size_t)c1 * P_DIM;
#pragma unroll
        for (int k = 0; k < P_DIM; k += 32) {
            FragB16 A0, A1, B0, B1;
            A0.q[0] = *(const uint4*)(a0 + k + kA);
            A0.q[1] = *(const uint4*)(a0 + k + kA + 16);
            A1.q[0] = *(const uint4*)(a1 + k + kA);
            A1.q[1] = *(const uint4*)(a1 + k + kA + 16);
            B0.q[0] = *(const uint4*)(b0 + k + kB);
            B0.q[1] = *(const uint4*)(b0 + k + kB + 16);
            B1.q[0] = *(const uint4*)(b1 + k + kB);
            B1.q[1] = *(const uint4*)(b1 + k + kB + 16);
            p00 = WMMA_BF16(A0.v, B0.v, p00);
            p01 = WMMA_BF16(A0.v, B1.v, p01);
            p10 = WMMA_BF16(A1.v, B0.v, p10);
            p11 = WMMA_BF16(A1.v, B1.v, p11);
        }
    }
    // S2 = Xn @ Xn^T 32x32 block, K = 256
    v8f e00 = {}, e01 = {}, e10 = {}, e11 = {};
    {
        const __bf16* a0 = Xn + (size_t)r0 * D_DIM;
        const __bf16* a1 = Xn + (size_t)r1 * D_DIM;
        const __bf16* b0 = Xn + (size_t)c0 * D_DIM;
        const __bf16* b1 = Xn + (size_t)c1 * D_DIM;
#pragma unroll
        for (int k = 0; k < D_DIM; k += 32) {
            FragB16 A0, A1, B0, B1;
            A0.q[0] = *(const uint4*)(a0 + k + kA);
            A0.q[1] = *(const uint4*)(a0 + k + kA + 16);
            A1.q[0] = *(const uint4*)(a1 + k + kA);
            A1.q[1] = *(const uint4*)(a1 + k + kA + 16);
            B0.q[0] = *(const uint4*)(b0 + k + kB);
            B0.q[1] = *(const uint4*)(b0 + k + kB + 16);
            B1.q[0] = *(const uint4*)(b1 + k + kB);
            B1.q[1] = *(const uint4*)(b1 + k + kB + 16);
            e00 = WMMA_BF16(A0.v, B0.v, e00);
            e01 = WMMA_BF16(A0.v, B1.v, e01);
            e10 = WMMA_BF16(A1.v, B0.v, e10);
            e11 = WMMA_BF16(A1.v, B1.v, e11);
        }
    }
    float s = 0.0f;
#pragma unroll
    for (int i = 0; i < 8; ++i) {
        float d;
        d = p00[i] - e00[i]; s += d * d;
        d = p01[i] - e01[i]; s += d * d;
        d = p10[i] - e10[i]; s += d * d;
        d = p11[i] - e11[i]; s += d * d;
    }
    s = waveReduce(s);
    const float f = (R == C) ? 1.0f : 2.0f;
    if (lane == 0)
        atomicAdd(out, s * f * (1.0f / ((float)N_NODES * (float)N_NODES)));
}

// --- Structure: fused  mean((A - Adj@A)^2).  HBM-bound on the 256 MB Adj
// stream (~11.7 us floor @ 23.3 TB/s).  8 waves/WG share a 16-row Adj slab
// (WGP$ reuse); Adj converted f32->bf16 in registers; prefetch streams ahead.
__global__ void __launch_bounds__(256)
structure_kernel(const float* __restrict__ Adj,
                 const float* __restrict__ A,
                 const __bf16* __restrict__ AbfT,
                 float* __restrict__ out) {
    const int lane = threadIdx.x & 31;
    const int wave = threadIdx.x >> 5;        // column tile 0..7
    const int rt   = blockIdx.x;              // row tile    0..511
    const int rowA = rt * 16 + (lane & 15);
    const int col  = wave * 16 + (lane & 15);
    const int kA = (lane < 16) ? 0 : 8;
    const int kB = (lane < 16) ? 0 : 8;

    const float*  adjRow = Adj  + (size_t)rowA * N_NODES;
    const __bf16* bcol   = AbfT + (size_t)col  * N_NODES;

    v8f c = {};
    for (int k = 0; k < N_NODES; k += 32) {
        if (k + 1024 < N_NODES)                 // stream Adj ahead of use
            __builtin_prefetch(adjRow + k + 1024 + kA, 0, 0);

        const float4 f0 = *(const float4*)(adjRow + k + kA);
        const float4 f1 = *(const float4*)(adjRow + k + kA + 4);
        const float4 f2 = *(const float4*)(adjRow + k + kA + 16);
        const float4 f3 = *(const float4*)(adjRow + k + kA + 20);
        const float tmp[16] = { f0.x, f0.y, f0.z, f0.w, f1.x, f1.y, f1.z, f1.w,
                                f2.x, f2.y, f2.z, f2.w, f3.x, f3.y, f3.z, f3.w };
        FragB16 a;
#pragma unroll
        for (int i = 0; i < 16; ++i) a.v[i] = (__bf16)tmp[i];

        FragB16 b;
        b.q[0] = *(const uint4*)(bcol + k + kB);
        b.q[1] = *(const uint4*)(bcol + k + kB + 16);

        c = WMMA_BF16(a.v, b.v, c);
    }
    // C/D layout: VGPR i -> M = i (lanes 0-15) or i+8 (lanes 16-31), N = lane%16
    float s = 0.0f;
#pragma unroll
    for (int i = 0; i < 8; ++i) {
        const int r = rt * 16 + i + ((lane < 16) ? 0 : 8);
        const float d = A[(size_t)r * P_DIM + col] - c[i];
        s += d * d;
    }
    s = waveReduce(s);
    if (lane == 0)
        atomicAdd(out, s * (1.0f / ((float)N_NODES * (float)P_DIM)));
}

extern "C" void kernel_launch(void* const* d_in, const int* in_sizes, int n_in,
                              void* d_out, int out_size, void* d_ws, size_t ws_size,
                              hipStream_t stream) {
    (void)in_sizes; (void)n_in; (void)out_size; (void)ws_size;
    const float* pred = (const float*)d_in[0];   // [N, P]
    const float* emb  = (const float*)d_in[1];   // [N, D]
    const float* adj  = (const float*)d_in[2];   // [N, N]
    const float* wts  = (const float*)d_in[3];   // [N, P]
    float* out = (float*)d_out;

    char* ws = (char*)d_ws;
    __bf16* Abf  = (__bf16*)(ws);                              // 2 MB
    __bf16* AbfT = (__bf16*)(ws + (size_t)2 * 1024 * 1024);    // 2 MB
    __bf16* Xn   = (__bf16*)(ws + (size_t)4 * 1024 * 1024);    // 4 MB

    init_out<<<1, 1, 0, stream>>>(out);
    normalize_embeddings<<<N_NODES / 8, 256, 0, stream>>>(emb, Xn);
    convert_and_weight<<<2048, 256, 0, stream>>>(pred, wts, Abf, AbfT, out);

    dim3 gridC(256 / 4, 256);   // 32x32 blocks, 4 waves/block along C
    coherence_kernel<<<gridC, 128, 0, stream>>>(Abf, Xn, out);
    structure_kernel<<<512, 256, 0, stream>>>(adj, pred, AbfT, out);
}
